// MEAM_824633721697
// MI455X (gfx1250) — matrix operand order repaired and checked by reading, not compiled, not run on hardware
//
#include <hip/hip_runtime.h>
#include <hip/hip_bf16.h>

typedef __attribute__((ext_vector_type(16))) _Float16 v16h;
typedef __attribute__((ext_vector_type(8)))  _Float16 v8h;
typedef __attribute__((ext_vector_type(8)))  float    v8f;

#define H_    128
#define HEADS_ 8
#define DH_   16
#define NB_   256          // batch B
#define NN_   200          // sequence n
#define NENC_ 5
#define EB_   (NB_*NENC_)  // 1280
#define NEGV_ (-1e9f)
#define EPS_  1e-5f

static inline long long CDIV(long long a, long long b){ return (a+b-1)/b; }

// ---------------------------------------------------------------------------
// CDNA5 async copy global->LDS (ASYNCcnt path) + wait helpers
// ---------------------------------------------------------------------------
__device__ __forceinline__ void async_b128(unsigned lds_off, const float* g) {
  asm volatile("global_load_async_to_lds_b128 %0, %1, off"
               :: "v"(lds_off), "v"(g) : "memory");
}
#define WAIT_ASYNC_0() asm volatile("s_wait_asynccnt 0" ::: "memory")
#define WAIT_ASYNC_8() asm volatile("s_wait_asynccnt 8" ::: "memory")

// ---------------------------------------------------------------------------
// WMMA fragment loads (wave32; contiguous k = 16*(lane>>4) + s in both A and B)
// f16-LDS variants (guarded GEMM): A row-major [64][32]h, B transposed [n][k]h
// ---------------------------------------------------------------------------
__device__ __forceinline__ v16h frag_a(const _Float16* a, int row0, int lane) {
  int r  = row0 + (lane & 15);
  int kb = lane >> 4;
  const _Float16* p = a + r * 32 + 16 * kb;
  v8h c0 = *(const v8h*)(p);
  v8h c1 = *(const v8h*)(p + 8);
  v16h f;
#pragma unroll
  for (int i = 0; i < 8; ++i) { f[i] = c0[i]; f[i + 8] = c1[i]; }
  return f;
}
__device__ __forceinline__ v16h frag_b(const _Float16* bt, int n0, int lane) {
  int nn = n0 + (lane & 15);
  int kb = lane >> 4;
  return *(const v16h*)(bt + nn * 32 + 16 * kb);
}
// f32-LDS variants (async GEMM): convert to f16 at fragment-load time.
__device__ __forceinline__ v16h frag_a32(const float* a, int row0, int lane) {
  int r  = row0 + (lane & 15);
  int kb = lane >> 4;
  const float* p = a + r * 32 + 16 * kb;
  v16h f;
#pragma unroll
  for (int i = 0; i < 8; ++i) { f[i] = (_Float16)p[i]; f[i + 8] = (_Float16)p[8 + i]; }
  return f;
}
__device__ __forceinline__ v16h frag_b32(const float* bs, int n0, int lane) {
  int nn = n0 + (lane & 15);
  int kb = lane >> 4;
  v16h f;
#pragma unroll
  for (int i = 0; i < 16; ++i) f[i] = (_Float16)bs[(kb * 16 + i) * 64 + nn];
  return f;
}

// ---------------------------------------------------------------------------
// Guarded batched GEMM (ragged tiles): C[z] = act(scale*A@op(B) + bias + res)
// A: MxK row-major. TRANSB=0: B KxN row-major. TRANSB=1: B NxK row-major.
// ---------------------------------------------------------------------------
template <int TRANSB>
__global__ __launch_bounds__(128) void k_gemm(
    const float* __restrict__ A, const float* __restrict__ Bm,
    float* __restrict__ C, int M, int N, int K,
    long long sA, long long sB, long long sC,
    const float* __restrict__ bias, const float* __restrict__ resid,
    float scale, int relu)
{
  __shared__ __align__(32) _Float16 As[64 * 32];
  __shared__ __align__(32) _Float16 Bs[64 * 32];
  const int tid  = threadIdx.x;
  const int lane = tid & 31;
  const int wv   = tid >> 5;
  const int wm   = wv >> 1, wn = wv & 1;
  const int row0 = blockIdx.y * 64;
  const int col0 = blockIdx.x * 64;
  const long long z = blockIdx.z;
  A  += z * sA;  Bm += z * sB;  C += z * sC;
  const float* R = resid ? (resid + z * sC) : nullptr;

  v8f acc[2][2] = {};

  for (int kt = 0; kt < K; kt += 32) {
    if (kt + 32 < K) {
      int pr = row0 + (tid >> 1);
      if (pr < M) __builtin_prefetch(&A[(long long)pr * K + kt + 32], 0, 1);
    }
    for (int i = tid; i < 64 * 32; i += 128) {
      int r = i >> 5, kk = i & 31;
      int gr = row0 + r, gk = kt + kk;
      float v = (gr < M && gk < K) ? A[(long long)gr * K + gk] : 0.f;
      As[i] = (_Float16)v;
    }
    if (TRANSB) {
      for (int i = tid; i < 64 * 32; i += 128) {
        int n = i >> 5, kk = i & 31;
        int gn = col0 + n, gk = kt + kk;
        float v = (gn < N && gk < K) ? Bm[(long long)gn * K + gk] : 0.f;
        Bs[n * 32 + kk] = (_Float16)v;
      }
    } else {
      for (int i = tid; i < 64 * 32; i += 128) {
        int n = i & 63, kk = i >> 6;
        int gn = col0 + n, gk = kt + kk;
        float v = (gn < N && gk < K) ? Bm[(long long)gk * N + gn] : 0.f;
        Bs[n * 32 + kk] = (_Float16)v;
      }
    }
    __syncthreads();

    v16h a0 = frag_a(As, wm * 32,      lane);
    v16h a1 = frag_a(As, wm * 32 + 16, lane);
    v16h b0 = frag_b(Bs, wn * 32,      lane);
    v16h b1 = frag_b(Bs, wn * 32 + 16, lane);
    acc[0][0] = __builtin_amdgcn_wmma_f32_16x16x32_f16(false, a0, false, b0, (short)0, acc[0][0], false, false);
    acc[0][1] = __builtin_amdgcn_wmma_f32_16x16x32_f16(false, a0, false, b1, (short)0, acc[0][1], false, false);
    acc[1][0] = __builtin_amdgcn_wmma_f32_16x16x32_f16(false, a1, false, b0, (short)0, acc[1][0], false, false);
    acc[1][1] = __builtin_amdgcn_wmma_f32_16x16x32_f16(false, a1, false, b1, (short)0, acc[1][1], false, false);
    __syncthreads();
  }

  int cn = lane & 15, kb = lane >> 4;
  for (int mi = 0; mi < 2; ++mi)
    for (int ni = 0; ni < 2; ++ni) {
      int r0 = row0 + wm * 32 + mi * 16;
      int c  = col0 + wn * 32 + ni * 16 + cn;
#pragma unroll
      for (int i = 0; i < 8; ++i) {
        int r = r0 + i + 8 * kb;
        if (r < M && c < N) {
          float v = acc[mi][ni][i] * scale;
          if (bias)  v += bias[c];
          if (R)     v += R[(long long)r * N + c];
          if (relu)  v = fmaxf(v, 0.f);
          C[(long long)r * N + c] = v;
        }
      }
    }
}

// ---------------------------------------------------------------------------
// Async double-buffered GEMM (NN only). Requires M%64==0, N%64==0, K%32==0.
// Raw f32 tiles staged with global_load_async_to_lds_b128 (8 issues/wave/tile,
// fixed count -> s_wait_asynccnt 8 retires the previous tile), f32->f16 cvt at
// fragment load, f32 WMMA accumulate.
// ---------------------------------------------------------------------------
__global__ __launch_bounds__(128) void k_gemm_async(
    const float* __restrict__ A, const float* __restrict__ Bm,
    float* __restrict__ C, int M, int N, int K,
    long long sA, long long sB, long long sC,
    const float* __restrict__ bias, const float* __restrict__ resid,
    float scale, int relu)
{
  __shared__ __align__(16) float As[2][64 * 32];   // [row][k]
  __shared__ __align__(16) float Bs[2][32 * 64];   // [k][n]
  const int tid  = threadIdx.x;
  const int lane = tid & 31;
  const int wv   = tid >> 5;
  const int wm   = wv >> 1, wn = wv & 1;
  const int row0 = blockIdx.y * 64;
  const int col0 = blockIdx.x * 64;
  const long long z = blockIdx.z;
  A  += z * sA;  Bm += z * sB;  C += z * sC;
  const float* R = resid ? (resid + z * sC) : nullptr;

  // stage one 64x32 A tile + 32x64 B tile into LDS buffer `buf` (async).
  auto stage = [&](int buf, int kt) {
    const float* Abase = A + (long long)row0 * K + kt;
#pragma unroll
    for (int j = 0; j < 4; ++j) {          // A: 512 chunks of 4 f32
      int c  = tid + j * 128;
      int r  = c >> 3, c4 = (c & 7) << 2;
      unsigned lo = (unsigned)(unsigned long long)&As[buf][r * 32 + c4];
      async_b128(lo, Abase + (long long)r * K + c4);
    }
#pragma unroll
    for (int j = 0; j < 4; ++j) {          // B: 512 chunks of 4 f32
      int c  = tid + j * 128;
      int kk = c >> 4, c4 = (c & 15) << 2;
      unsigned lo = (unsigned)(unsigned long long)&Bs[buf][kk * 64 + c4];
      async_b128(lo, Bm + (long long)(kt + kk) * N + col0 + c4);
    }
  };

  v8f acc[2][2] = {};
  stage(0, 0);
  int i = 0;
  for (int kt = 0; kt < K; kt += 32, ++i) {
    if (kt + 32 < K) { stage((i + 1) & 1, kt + 32); WAIT_ASYNC_8(); }
    else             { WAIT_ASYNC_0(); }
    __syncthreads();

    const float* Ab = As[i & 1];
    const float* Bb = Bs[i & 1];
    v16h a0 = frag_a32(Ab, wm * 32,      lane);
    v16h a1 = frag_a32(Ab, wm * 32 + 16, lane);
    v16h b0 = frag_b32(Bb, wn * 32,      lane);
    v16h b1 = frag_b32(Bb, wn * 32 + 16, lane);
    acc[0][0] = __builtin_amdgcn_wmma_f32_16x16x32_f16(false, a0, false, b0, (short)0, acc[0][0], false, false);
    acc[0][1] = __builtin_amdgcn_wmma_f32_16x16x32_f16(false, a0, false, b1, (short)0, acc[0][1], false, false);
    acc[1][0] = __builtin_amdgcn_wmma_f32_16x16x32_f16(false, a1, false, b0, (short)0, acc[1][0], false, false);
    acc[1][1] = __builtin_amdgcn_wmma_f32_16x16x32_f16(false, a1, false, b1, (short)0, acc[1][1], false, false);
    __syncthreads();   // all waves done reading buf[i&1] before it is re-staged
  }

  int cn = lane & 15, kb = lane >> 4;
  for (int mi = 0; mi < 2; ++mi)
    for (int ni = 0; ni < 2; ++ni) {
      int r0 = row0 + wm * 32 + mi * 16;
      int c  = col0 + wn * 32 + ni * 16 + cn;
#pragma unroll
      for (int ii = 0; ii < 8; ++ii) {
        int r = r0 + ii + 8 * kb;
        float v = acc[mi][ni][ii] * scale;
        if (bias)  v += bias[c];
        if (R)     v += R[(long long)r * N + c];
        if (relu)  v = fmaxf(v, 0.f);
        C[(long long)r * N + c] = v;
      }
    }
}

// ---------------------------------------------------------------------------
// small elementwise / reduction kernels
// ---------------------------------------------------------------------------
__global__ void k_fill(float* p, float v, long long n) {
  long long i = (long long)blockIdx.x * blockDim.x + threadIdx.x;
  if (i < n) p[i] = v;
}

__global__ void k_pos(float* pos) {   // sinusoidal table for 2 positions x 128
  int t = threadIdx.x;
  if (t >= 256) return;
  int p = t >> 7, i = t & 127;
  float e   = (2.0f * (float)(i >> 1)) / 128.0f;
  float ang = (float)p / powf(10000.f, e);
  pos[t] = (i & 1) ? cosf(ang) : sinf(ang);
}

__global__ void k_embed(const float* __restrict__ x, const float* __restrict__ W,
                        const float* __restrict__ b, float* __restrict__ h, long long M) {
  long long id = (long long)blockIdx.x * blockDim.x + threadIdx.x;
  if (id >= M * 128) return;
  long long r = id >> 7; int f = (int)(id & 127);
  h[id] = x[r * 2] * W[f] + x[r * 2 + 1] * W[128 + f] + b[f];
}

__global__ void k_split_heads(const float* __restrict__ X, float* __restrict__ Y,
                              long long rows, int n) {
  long long id = (long long)blockIdx.x * blockDim.x + threadIdx.x;
  if (id >= rows * 128) return;
  long long r = id >> 7; int c = (int)(id & 127);
  int h = c >> 4, d = c & 15;
  long long b = r / n; int j = (int)(r % n);
  Y[(((b * 8 + h) * (long long)n + j) << 4) + d] = X[id];
}

__global__ void k_merge_heads(const float* __restrict__ Y, float* __restrict__ X,
                              long long rows, int n) {
  long long id = (long long)blockIdx.x * blockDim.x + threadIdx.x;
  if (id >= rows * 128) return;
  long long r = id >> 7; int c = (int)(id & 127);
  int h = c >> 4, d = c & 15;
  long long b = r / n; int j = (int)(r % n);
  X[id] = Y[(((b * 8 + h) * (long long)n + j) << 4) + d];
}

__global__ __launch_bounds__(256) void k_softmax_rows(float* S, long long rows, int nc) {
  long long row = (long long)blockIdx.x * 8 + (threadIdx.x >> 5);
  int lane = threadIdx.x & 31;
  if (row >= rows) return;
  float* p = S + row * (long long)nc;
  float m = -3.4e38f;
  for (int j = lane; j < nc; j += 32) m = fmaxf(m, p[j]);
  for (int o = 16; o > 0; o >>= 1)   m = fmaxf(m, __shfl_xor(m, o, 32));
  float s = 0.f;
  for (int j = lane; j < nc; j += 32) { float e = __expf(p[j] - m); p[j] = e; s += e; }
  for (int o = 16; o > 0; o >>= 1)   s += __shfl_xor(s, o, 32);
  float inv = 1.f / s;
  for (int j = lane; j < nc; j += 32) p[j] *= inv;
}

__global__ __launch_bounds__(256) void k_bn_stats(const float* __restrict__ X,
                                                  float* __restrict__ mv, long long M) {
  __shared__ float s1[256], s2[256];
  int f = blockIdx.x, tid = threadIdx.x;
  float a = 0.f, b = 0.f;
  for (long long r = tid; r < M; r += 256) { float v = X[r * 128 + f]; a += v; b += v * v; }
  s1[tid] = a; s2[tid] = b; __syncthreads();
  for (int o = 128; o > 0; o >>= 1) {
    if (tid < o) { s1[tid] += s1[tid + o]; s2[tid] += s2[tid + o]; }
    __syncthreads();
  }
  if (tid == 0) { float mu = s1[0] / (float)M; mv[f] = mu; mv[128 + f] = s2[0] / (float)M - mu * mu; }
}

__global__ void k_bn_apply(const float* __restrict__ X, float* __restrict__ Y,
                           const float* __restrict__ mv, const float* __restrict__ g,
                           const float* __restrict__ b, long long M) {
  long long id = (long long)blockIdx.x * blockDim.x + threadIdx.x;
  if (id >= M * 128) return;
  int f = (int)(id & 127);
  Y[id] = g[f] * (X[id] - mv[f]) * rsqrtf(mv[128 + f] + EPS_) + b[f];
}

// T=2 self-attention for the decoder (per eb x head)
__global__ void k_attn2(const float* __restrict__ Q, const float* __restrict__ K,
                        const float* __restrict__ V, float* __restrict__ O, int EB) {
  int id = blockIdx.x * blockDim.x + threadIdx.x;
  if (id >= EB * 8) return;
  int eb = id >> 3, h = id & 7;
  long long base = (long long)eb * 256 + h * 16;
  const float *q0 = Q + base, *q1 = Q + base + 128;
  const float *k0 = K + base, *k1 = K + base + 128;
  const float *v0 = V + base, *v1 = V + base + 128;
  float s00 = 0, s01 = 0, s10 = 0, s11 = 0;
#pragma unroll
  for (int d = 0; d < 16; ++d) {
    s00 += q0[d] * k0[d]; s01 += q0[d] * k1[d];
    s10 += q1[d] * k0[d]; s11 += q1[d] * k1[d];
  }
  s00 *= 0.25f; s01 *= 0.25f; s10 *= 0.25f; s11 *= 0.25f;
  float m0 = fmaxf(s00, s01), m1 = fmaxf(s10, s11);
  float e00 = __expf(s00 - m0), e01 = __expf(s01 - m0);
  float e10 = __expf(s10 - m1), e11 = __expf(s11 - m1);
  float i0 = 1.f / (e00 + e01), i1 = 1.f / (e10 + e11);
#pragma unroll
  for (int d = 0; d < 16; ++d) {
    O[base + d]       = (e00 * v0[d] + e01 * v1[d]) * i0;
    O[base + 128 + d] = (e10 * v0[d] + e11 * v1[d]) * i1;
  }
}

__global__ void k_inorm(const float* __restrict__ X, float* __restrict__ Y,
                        const float* __restrict__ g, const float* __restrict__ b, int EB) {
  int id = blockIdx.x * blockDim.x + threadIdx.x;
  if (id >= EB * 128) return;
  int eb = id >> 7, f = id & 127;
  long long base = (long long)eb * 256 + f;
  float a = X[base], c = X[base + 128];
  float mu  = 0.5f * (a + c);
  float var = 0.5f * ((a - mu) * (a - mu) + (c - mu) * (c - mu));
  float inv = rsqrtf(var + EPS_);
  Y[base]       = g[f] * (a - mu) * inv + b[f];
  Y[base + 128] = g[f] * (c - mu) * inv + b[f];
}

__global__ void k_addpos(const float* __restrict__ D, const float* __restrict__ pos,
                         float* __restrict__ Y, int EB) {
  int id = blockIdx.x * blockDim.x + threadIdx.x;
  if (id >= EB * 256) return;
  Y[id] = D[id] + pos[id & 255];
}

// masked glimpse attention, one block per (eb, head), 64 threads
__global__ __launch_bounds__(64) void k_glimpse(
    const float* __restrict__ D, const float* __restrict__ gK,
    const float* __restrict__ gV, const float* __restrict__ mask,
    float* __restrict__ Out, int n) {
  __shared__ float sred[64];
  __shared__ float mrow[2], srow[2];
  __shared__ float osum[32];
  int eb = blockIdx.x >> 3, h = blockIdx.x & 7;
  int tid = threadIdx.x;
  const float* qp = D + (long long)eb * 256 + h * 16;
  float q0[16], q1[16];
#pragma unroll
  for (int d = 0; d < 16; ++d) { q0[d] = qp[d]; q1[d] = qp[128 + d]; }
  float sl0[4], sl1[4], lm0 = -3.4e38f, lm1 = -3.4e38f;
  for (int kk = 0; kk < 4; ++kk) {
    int j = tid + kk * 64;
    if (j < n) {
      const float* kp = gK + ((long long)eb * n + j) * 128 + h * 16;
      float d0 = 0, d1 = 0;
#pragma unroll
      for (int d = 0; d < 16; ++d) { float kv = kp[d]; d0 += q0[d] * kv; d1 += q1[d] * kv; }
      float mb = (mask[(long long)eb * n + j] > 0.f) ? 0.f : NEGV_;
      sl0[kk] = d0 * 0.25f + mb; sl1[kk] = d1 * 0.25f + mb;
      lm0 = fmaxf(lm0, sl0[kk]); lm1 = fmaxf(lm1, sl1[kk]);
    } else { sl0[kk] = -3.4e38f; sl1[kk] = -3.4e38f; }
  }
  sred[tid] = lm0; __syncthreads();
  for (int o = 32; o > 0; o >>= 1) { if (tid < o) sred[tid] = fmaxf(sred[tid], sred[tid + o]); __syncthreads(); }
  if (tid == 0) mrow[0] = sred[0];
  __syncthreads();
  sred[tid] = lm1; __syncthreads();
  for (int o = 32; o > 0; o >>= 1) { if (tid < o) sred[tid] = fmaxf(sred[tid], sred[tid + o]); __syncthreads(); }
  if (tid == 0) mrow[1] = sred[0];
  __syncthreads();
  float ex0[4], ex1[4], ls0 = 0.f, ls1 = 0.f;
  for (int kk = 0; kk < 4; ++kk) {
    int j = tid + kk * 64;
    if (j < n) {
      ex0[kk] = __expf(sl0[kk] - mrow[0]); ls0 += ex0[kk];
      ex1[kk] = __expf(sl1[kk] - mrow[1]); ls1 += ex1[kk];
    } else { ex0[kk] = 0.f; ex1[kk] = 0.f; }
  }
  sred[tid] = ls0; __syncthreads();
  for (int o = 32; o > 0; o >>= 1) { if (tid < o) sred[tid] += sred[tid + o]; __syncthreads(); }
  if (tid == 0) srow[0] = sred[0];
  __syncthreads();
  sred[tid] = ls1; __syncthreads();
  for (int o = 32; o > 0; o >>= 1) { if (tid < o) sred[tid] += sred[tid + o]; __syncthreads(); }
  if (tid == 0) srow[1] = sred[0];
  if (tid < 32) osum[tid] = 0.f;
  __syncthreads();
  float oa0[16] = {}, oa1[16] = {};
  for (int kk = 0; kk < 4; ++kk) {
    int j = tid + kk * 64;
    if (j < n) {
      const float* vp = gV + ((long long)eb * n + j) * 128 + h * 16;
#pragma unroll
      for (int d = 0; d < 16; ++d) { float vv = vp[d]; oa0[d] += ex0[kk] * vv; oa1[d] += ex1[kk] * vv; }
    }
  }
#pragma unroll
  for (int d = 0; d < 16; ++d) { atomicAdd(&osum[d], oa0[d]); atomicAdd(&osum[16 + d], oa1[d]); }
  __syncthreads();
  if (tid < 32) {
    int tq = tid >> 4, d = tid & 15;
    Out[((long long)eb * 2 + tq) * 128 + h * 16 + d] = osum[tid] / srow[tq];
  }
}

__global__ void k_sumT(const float* __restrict__ D, float* __restrict__ Q, int EB) {
  int id = blockIdx.x * blockDim.x + threadIdx.x;
  if (id >= EB * 128) return;
  int eb = id >> 7, f = id & 127;
  Q[id] = D[(long long)eb * 256 + f] + D[(long long)eb * 256 + 128 + f];
}

__global__ void k_logits(const float* __restrict__ QV, const float* __restrict__ PK,
                         float* __restrict__ U, int n) {
  long long id = (long long)blockIdx.x * blockDim.x + threadIdx.x;
  if (id >= (long long)EB_ * n) return;
  int eb = (int)(id / n), j = (int)(id % n);
  const float* a = QV + (long long)eb * 128;
  const float* k = PK + ((long long)eb * n + j) * 128;
  float s = 0.f;
  for (int d = 0; d < 128; ++d) s += a[d] * k[d];
  U[id] = 10.f * tanhf(s * 0.08838834764831845f);  // 1/sqrt(128), then clip
}

// mask / top-k(10) / softmax / argmax / bookkeeping  (one thread per eb row)
__global__ void k_select(const float* __restrict__ U, float* __restrict__ mask,
                         float* __restrict__ routes, float* __restrict__ probsO,
                         float* __restrict__ firstO, int* __restrict__ idxB,
                         int* __restrict__ desB, int step, int n) {
  int eb = blockIdx.x * blockDim.x + threadIdx.x;
  if (eb >= EB_) return;
  float f[NN_];
  unsigned char sel[NN_];
  const float* up = U + (long long)eb * n;
  float* mp = mask + (long long)eb * n;
  for (int j = 0; j < n; ++j) { f[j] = up[j] + (mp[j] > 0.f ? 0.f : NEGV_); sel[j] = 0; }
  if (step > 0) {
    for (int t = 0; t < 10; ++t) {           // top-k over masked_u, first-occurrence ties
      float best = -3.4e38f; int bi = 0;
      for (int j = 0; j < n; ++j) if (!sel[j] && f[j] > best) { best = f[j]; bi = j; }
      sel[bi] = 1;
    }
    for (int j = 0; j < n; ++j) f[j] = up[j] + ((mp[j] > 0.f && sel[j]) ? 0.f : NEGV_);
  }
  float m = -3.4e38f; int idx = 0;
  for (int j = 0; j < n; ++j) if (f[j] > m) { m = f[j]; idx = j; }
  float s = 0.f;
  for (int j = 0; j < n; ++j) { f[j] = __expf(f[j] - m); s += f[j]; }
  float inv = 1.f / s;
  float chosen = f[idx] * inv;
  if (step == 0) {
    desB[eb] = idx;
    for (int j = 0; j < n; ++j) firstO[(long long)eb * n + j] = f[j] * inv;
  }
  idxB[eb] = idx;
  mp[idx] = 0.f;
  routes[(long long)eb * n + step] = (float)idx;
  probsO[(long long)eb * n + step] = chosen;
}

__global__ void k_build_dce(const float* __restrict__ NE, const int* __restrict__ idxB,
                            const int* __restrict__ desB, float* __restrict__ dce, int n) {
  int id = blockIdx.x * blockDim.x + threadIdx.x;
  if (id >= EB_ * 128) return;
  int eb = id >> 7, fch = id & 127;
  int di = desB[eb], ci = idxB[eb];
  dce[(long long)eb * 256 + fch]       = NE[((long long)eb * n + di) * 128 + fch];
  dce[(long long)eb * 256 + 128 + fch] = NE[((long long)eb * n + ci) * 128 + fch];
}

__global__ void k_init_dce(const float* __restrict__ desH, const float* __restrict__ curH,
                           float* __restrict__ dce) {
  int id = blockIdx.x * blockDim.x + threadIdx.x;
  if (id >= EB_ * 128) return;
  int eb = id >> 7, f = id & 127;
  dce[(long long)eb * 256 + f]       = desH[f];
  dce[(long long)eb * 256 + 128 + f] = curH[f];
}

// ---------------------------------------------------------------------------
// host orchestration
// ---------------------------------------------------------------------------
struct Lay {
  const float *Wq, *Wk, *Wv, *Wo, *g1, *b1n, *W1, *b1, *W2, *b2, *g2, *b2n;
};

extern "C" void kernel_launch(void* const* d_in, const int* in_sizes, int n_in,
                              void* d_out, int out_size, void* d_ws, size_t ws_size,
                              hipStream_t stream) {
  (void)in_sizes; (void)n_in; (void)out_size; (void)ws_size;

  // ---- input walker, setup_inputs() insertion order ----
  int pi = 0;
  auto NXT = [&]() { return (const float*)d_in[pi++]; };
  const float* x_all = NXT();
  const float* cur_h = NXT();
  const float* des_h = NXT();
  const float* encW[NENC_]; const float* encB[NENC_]; Lay encL[NENC_][3];
  auto rdL = [&](Lay& L) {
    L.Wq = NXT(); L.Wk = NXT(); L.Wv = NXT(); L.Wo = NXT();
    L.g1 = NXT(); L.b1n = NXT();
    L.W1 = NXT(); L.b1 = NXT(); L.W2 = NXT(); L.b2 = NXT();
    L.g2 = NXT(); L.b2n = NXT();
  };
  for (int e = 0; e < NENC_; ++e) {
    encW[e] = NXT(); encB[e] = NXT();
    for (int l = 0; l < 3; ++l) rdL(encL[e][l]);
  }
  const float* gKw[2] = { NXT(), NXT() };
  const float* gVw[2] = { NXT(), NXT() };
  const float* gVO[2] = { NXT(), NXT() };
  Lay decL[2]; rdL(decL[0]); rdL(decL[1]);
  const float* probQ = NXT();
  const float* probK = NXT();

  // ---- workspace carve ----
  float* wsf = (float*)d_ws;
  size_t off = 0;
  auto AL = [&](long long nel) { float* p = wsf + off; off += (size_t)nel; return p; };
  const long long Mtok = (long long)NB_ * NN_;      // 51200 tokens per encoder
  const long long Meb  = (long long)EB_ * NN_;      // 256000 rows
  float* node_emb = AL(5LL * Mtok * 128);
  float* gKb[2] = { AL(Meb * 128), AL(Meb * 128) };
  float* gVb[2] = { AL(Meb * 128), AL(Meb * 128) };
  float* probKey = AL(Meb * 128);
  float* Qb = AL(Mtok * 128); float* Kb2 = AL(Mtok * 128); float* Vb2 = AL(Mtok * 128);
  float* Qh = AL(Mtok * 128); float* Kh  = AL(Mtok * 128); float* Vh  = AL(Mtok * 128);
  float* Sb = AL(2048LL * NN_ * NN_);
  float* Oh = AL(Mtok * 128);
  float* Om = AL(Mtok * 128);
  float* T1 = AL(Mtok * 128);
  float* Ff = AL(Mtok * 512);
  float* mv = AL(256);
  float* pos = AL(256);
  float* mask = AL(Meb);
  float* dceC = AL(EB_ * 256); float* W0 = AL(EB_ * 256);
  float* TA = AL(EB_ * 256);   float* TB = AL(EB_ * 256);
  float* Qd = AL(EB_ * 256);   float* Kd = AL(EB_ * 256); float* Vd = AL(EB_ * 256);
  float* O2 = AL(EB_ * 256);   float* GB = AL(EB_ * 256);
  float* MIDb = AL(2LL * EB_ * 512);
  float* QS = AL(EB_ * 128);   float* QV = AL(EB_ * 128);
  float* Ub = AL(Meb);
  int* idxB = (int*)AL(EB_);
  int* desB = (int*)AL(EB_);

  float* routes_o = (float*)d_out;
  float* probs_o  = routes_o + Meb;
  float* first_o  = probs_o + Meb;

  auto gemm = [&](const float* A, const float* Bm, float* C, int M, int N, int K,
                  int batch, long long sA, long long sB, long long sC,
                  const float* bias, const float* resid, float scale, int relu, int transb) {
    dim3 g((unsigned)CDIV(N, 64), (unsigned)CDIV(M, 64), (unsigned)batch), bl(128);
    if (!transb && (M % 64 == 0) && (N % 64 == 0) && (K % 32 == 0))
      k_gemm_async<<<g, bl, 0, stream>>>(A, Bm, C, M, N, K, sA, sB, sC, bias, resid, scale, relu);
    else if (transb)
      k_gemm<1><<<g, bl, 0, stream>>>(A, Bm, C, M, N, K, sA, sB, sC, bias, resid, scale, relu);
    else
      k_gemm<0><<<g, bl, 0, stream>>>(A, Bm, C, M, N, K, sA, sB, sC, bias, resid, scale, relu);
  };

  // ---- init ----
  k_pos<<<1, 256, 0, stream>>>(pos);
  k_fill<<<(unsigned)CDIV(Meb, 256), 256, 0, stream>>>(mask, 1.f, Meb);
  k_init_dce<<<(unsigned)CDIV(EB_ * 128, 256), 256, 0, stream>>>(des_h, cur_h, dceC);

  // ---- 5 encoders, 3 layers each ----
  for (int e = 0; e < NENC_; ++e) {
    float* h = node_emb + (long long)e * Mtok * 128;
    k_embed<<<(unsigned)CDIV(Mtok * 128, 256), 256, 0, stream>>>(x_all, encW[e], encB[e], h, Mtok);
    for (int l = 0; l < 3; ++l) {
      Lay& L = encL[e][l];
      gemm(h, L.Wq, Qb,  (int)Mtok, 128, 128, 1, 0, 0, 0, nullptr, nullptr, 1.f, 0, 0);
      gemm(h, L.Wk, Kb2, (int)Mtok, 128, 128, 1, 0, 0, 0, nullptr, nullptr, 1.f, 0, 0);
      gemm(h, L.Wv, Vb2, (int)Mtok, 128, 128, 1, 0, 0, 0, nullptr, nullptr, 1.f, 0, 0);
      k_split_heads<<<(unsigned)CDIV(Mtok * 128, 256), 256, 0, stream>>>(Qb, Qh, Mtok, NN_);
      k_split_heads<<<(unsigned)CDIV(Mtok * 128, 256), 256, 0, stream>>>(Kb2, Kh, Mtok, NN_);
      k_split_heads<<<(unsigned)CDIV(Mtok * 128, 256), 256, 0, stream>>>(Vb2, Vh, Mtok, NN_);
      // S = Qh @ Kh^T / sqrt(16), batched over B*heads = 2048  (ragged -> guarded NT)
      gemm(Qh, Kh, Sb, NN_, NN_, DH_, NB_ * HEADS_,
           (long long)NN_ * DH_, (long long)NN_ * DH_, (long long)NN_ * NN_,
           nullptr, nullptr, 0.25f, 0, 1);
      k_softmax_rows<<<(unsigned)CDIV(2048LL * NN_, 8), 256, 0, stream>>>(Sb, 2048LL * NN_, NN_);
      // Oh = S @ Vh  (ragged -> guarded NN)
      gemm(Sb, Vh, Oh, NN_, DH_, NN_, NB_ * HEADS_,
           (long long)NN_ * NN_, (long long)NN_ * DH_, (long long)NN_ * DH_,
           nullptr, nullptr, 1.f, 0, 0);
      k_merge_heads<<<(unsigned)CDIV(Mtok * 128, 256), 256, 0, stream>>>(Oh, Om, Mtok, NN_);
      gemm(Om, L.Wo, T1, (int)Mtok, 128, 128, 1, 0, 0, 0, nullptr, h, 1.f, 0, 0);
      k_bn_stats<<<128, 256, 0, stream>>>(T1, mv, Mtok);
      k_bn_apply<<<(unsigned)CDIV(Mtok * 128, 256), 256, 0, stream>>>(T1, h, mv, L.g1, L.b1n, Mtok);
      gemm(h, L.W1, Ff, (int)Mtok, 512, 128, 1, 0, 0, 0, L.b1, nullptr, 1.f, 1, 0);
      gemm(Ff, L.W2, T1, (int)Mtok, 128, 512, 1, 0, 0, 0, L.b2, h, 1.f, 0, 0);
      k_bn_stats<<<128, 256, 0, stream>>>(T1, mv, Mtok);
      k_bn_apply<<<(unsigned)CDIV(Mtok * 128, 256), 256, 0, stream>>>(T1, h, mv, L.g2, L.b2n, Mtok);
    }
  }

  // ---- glimpse keys/values + prob keys ----
  for (int i = 0; i < 2; ++i) {
    gemm(node_emb, gKw[i], gKb[i], (int)Meb, 128, 128, 1, 0, 0, 0, nullptr, nullptr, 1.f, 0, 0);
    gemm(node_emb, gVw[i], gVb[i], (int)Meb, 128, 128, 1, 0, 0, 0, nullptr, nullptr, 1.f, 0, 0);
  }
  gemm(node_emb, probK, probKey, (int)Meb, 128, 128, 1, 0, 0, 0, nullptr, nullptr, 1.f, 0, 0);

  // ---- 200 sequential decode steps ----
  for (int step = 0; step < NN_; ++step) {
    k_addpos<<<(unsigned)CDIV(EB_ * 256, 256), 256, 0, stream>>>(dceC, pos, W0, EB_);
    for (int i = 0; i < 2; ++i) {
      Lay& L = decL[i];
      gemm(W0, L.Wq, Qd, EB_ * 2, 128, 128, 1, 0, 0, 0, nullptr, nullptr, 1.f, 0, 0);
      gemm(W0, L.Wk, Kd, EB_ * 2, 128, 128, 1, 0, 0, 0, nullptr, nullptr, 1.f, 0, 0);
      gemm(W0, L.Wv, Vd, EB_ * 2, 128, 128, 1, 0, 0, 0, nullptr, nullptr, 1.f, 0, 0);
      k_attn2<<<(unsigned)CDIV(EB_ * 8, 256), 256, 0, stream>>>(Qd, Kd, Vd, O2, EB_);
      gemm(O2, L.Wo, TA, EB_ * 2, 128, 128, 1, 0, 0, 0, nullptr, W0, 1.f, 0, 0);
      k_inorm<<<(unsigned)CDIV(EB_ * 128, 256), 256, 0, stream>>>(TA, TA, L.g1, L.b1n, EB_);
      gemm(TA, L.W1, MIDb, EB_ * 2, 512, 128, 1, 0, 0, 0, L.b1, nullptr, 1.f, 1, 0);
      gemm(MIDb, L.W2, TB, EB_ * 2, 128, 512, 1, 0, 0, 0, L.b2, TA, 1.f, 0, 0);
      k_inorm<<<(unsigned)CDIV(EB_ * 128, 256), 256, 0, stream>>>(TB, W0, L.g2, L.b2n, EB_);
      k_glimpse<<<EB_ * 8, 64, 0, stream>>>(W0, gKb[i], gVb[i], mask, GB, NN_);
      gemm(GB, gVO[i], W0, EB_ * 2, 128, 128, 1, 0, 0, 0, nullptr, nullptr, 1.f, 0, 0);
    }
    k_sumT<<<(unsigned)CDIV(EB_ * 128, 256), 256, 0, stream>>>(W0, QS, EB_);
    gemm(QS, probQ, QV, EB_, 128, 128, 1, 0, 0, 0, nullptr, nullptr, 1.f, 0, 0);
    k_logits<<<(unsigned)CDIV(Meb, 256), 256, 0, stream>>>(QV, probKey, Ub, NN_);
    k_select<<<(unsigned)CDIV(EB_, 128), 128, 0, stream>>>(Ub, mask, routes_o, probs_o,
                                                           first_o, idxB, desB, step, NN_);
    k_build_dce<<<(unsigned)CDIV(EB_ * 128, 256), 256, 0, stream>>>(node_emb, idxB, desB, dceC, NN_);
  }
}
